// Model_3710851744524
// MI455X (gfx1250) — matrix operand order repaired and checked by reading, not compiled, not run on hardware
//
#include <hip/hip_runtime.h>
#include <cstdint>
#include <cstddef>

// ---------------- model constants ----------------
#define BB   16
#define SSL  512
#define PPL  256
#define LL2  768      // S + P
#define CCH  256
#define EEL  2
#define FGN  4
#define GSN  8
#define VV1  257      // S/2 + 1   (true rfft bins, pass 1)
#define VV2  385      // (S+P)/2+1 (true rfft bins, pass 2)
#define VP1  320      // padded bins (mult of 64); pad rows are exact zeros
#define VP2  448
#define VGMAX 97

typedef __attribute__((ext_vector_type(16))) _Float16 v16h;
typedef __attribute__((ext_vector_type(8)))  _Float16 v8h;
typedef __attribute__((ext_vector_type(8)))  float    v8f;

// =====================================================================
// WMMA GEMM:  C[M,N] = beta*C + alpha * A[M,K] x B    (f16 in, f32 out)
//   A row-major, lda == K.  CALLER CONTRACT (all met by this launcher):
//     * K % 32 == 0, N % 64 == 0
//     * A has at least ceil(M/64)*64 readable rows per batch slab
//     * nk-B (W[N,K]) has exactly N rows; kn-B ([K,N]) has K rows
//   => all tile loads are unconditional aligned v8h; only C store masks M.
//   Block tile 64x64, 256 threads = 8 waves; each wave -> 16x32 strip
//   (2 x v_wmma_f32_16x16x32_f16 per K-step sharing one A fragment).
// =====================================================================
__global__ __launch_bounds__(256)
void gemm_f16_wmma(const _Float16* __restrict__ Am, long strideA,
                   const _Float16* __restrict__ Bm, long strideB,
                   float* __restrict__ Cm, long strideC,
                   int M, int N, int K, int b_kn,
                   const float* __restrict__ alpha_ptr, float beta,
                   const float* __restrict__ bias, int bias_mode)
{
    __shared__ _Float16 As[64][40];   // [m][k], 80B row stride (16B aligned)
    __shared__ _Float16 Bs[64][40];   // [n][k]

    Am += (size_t)blockIdx.z * (size_t)strideA;
    Bm += (size_t)blockIdx.z * (size_t)strideB;
    Cm += (size_t)blockIdx.z * (size_t)strideC;

    const int tid  = threadIdx.x;
    const int lane = tid & 31;
    const int wave = tid >> 5;
    const int wm = (wave >> 1) << 4;          // 0,16,32,48  wave row offset
    const int wn = (wave & 1) << 5;           // 0 or 32     wave col offset
    const int blockM = blockIdx.y << 6;
    const int blockN = blockIdx.x << 6;

    const int lr  = tid >> 2;                 // 0..63  loader row (A / nk-B)
    const int lc8 = (tid & 3) << 3;           // 0,8,16,24 loader k chunk
    const int kr  = tid >> 3;                 // 0..31  loader k row (kn-B)
    const int kn8 = (tid & 7) << 3;           // 0..56  loader n chunk

    v8f acc0 = {}, acc1 = {};
    const int kTiles = K >> 5;

    for (int kt = 0; kt < kTiles; ++kt) {
        const int k0 = kt << 5;
        // ---- stage A tile (unconditional aligned load) ----
        {
            const int gm = blockM + lr;
            const int gk = k0 + lc8;
            v8h t = *(const v8h*)(Am + (size_t)gm * K + gk);
            *(v8h*)(&As[lr][lc8]) = t;
        }
        // ---- stage B tile into Bs[n][k] ----
        if (b_kn) {
            const int gk = k0 + kr;
            const int gn = blockN + kn8;
            v8h t = *(const v8h*)(Bm + (size_t)gk * N + gn);
#pragma unroll
            for (int j = 0; j < 8; ++j) Bs[kn8 + j][kr] = t[j];  // transpose
        } else {
            const int gn = blockN + lr;
            const int gk = k0 + lc8;
            v8h t = *(const v8h*)(Bm + (size_t)gn * K + gk);
            *(v8h*)(&Bs[lr][lc8]) = t;
        }
        __syncthreads();

        // ---- fragments per CDNA5 WMMA layouts ----
        // A 16x32 f16: lane<16 -> K 0..7 & 16..23 ; lane>=16 -> K 8..15 & 24..31
        const int am = wm + (lane & 15);
        const int kb = (lane >> 4) << 3;
        v8h a0 = *(const v8h*)(&As[am][kb]);
        v8h a1 = *(const v8h*)(&As[am][kb + 16]);
        v16h av = __builtin_shufflevector(a0, a1,
                    0,1,2,3,4,5,6,7,8,9,10,11,12,13,14,15);
        // B 32x16 f16: lane<16 -> K 0..7 & 16..23 ; lane>=16 -> K 8..15 & 24..31
        const int bn0 = wn + (lane & 15);
        const int kbb = (lane >> 4) << 3;
        v8h b0 = *(const v8h*)(&Bs[bn0][kbb]);
        v8h b1 = *(const v8h*)(&Bs[bn0][kbb + 16]);
        v16h bv0 = __builtin_shufflevector(b0, b1,
                    0,1,2,3,4,5,6,7,8,9,10,11,12,13,14,15);
        v8h b2 = *(const v8h*)(&Bs[bn0 + 16][kbb]);
        v8h b3 = *(const v8h*)(&Bs[bn0 + 16][kbb + 16]);
        v16h bv1 = __builtin_shufflevector(b2, b3,
                    0,1,2,3,4,5,6,7,8,9,10,11,12,13,14,15);

        acc0 = __builtin_amdgcn_wmma_f32_16x16x32_f16(
                   false, av, false, bv0, (short)0, acc0, false, false);
        acc1 = __builtin_amdgcn_wmma_f32_16x16x32_f16(
                   false, av, false, bv1, (short)0, acc1, false, false);
        __syncthreads();
    }

    const float alpha = alpha_ptr ? *alpha_ptr : 1.0f;
    // C 16x16 f32: lane%16 = N, VGPR r -> M = r + 8*(lane/16)
    const int gn0   = blockN + wn + (lane & 15);
    const int mbase = blockM + wm + ((lane >> 4) << 3);
#pragma unroll
    for (int rr = 0; rr < 8; ++rr) {
        const int gm = mbase + rr;
        if (gm < M) {
            {
                float v = alpha * acc0[rr];
                if (bias) v += (bias_mode == 1) ? bias[gn0] : bias[gm];
                float* p = Cm + (size_t)gm * N + gn0;
                if (beta != 0.0f) v = beta * (*p) + v;
                *p = v;
            }
            {
                float v = alpha * acc1[rr];
                if (bias) v += (bias_mode == 1) ? bias[gn0 + 16] : bias[gm];
                float* p = Cm + (size_t)gm * N + gn0 + 16;
                if (beta != 0.0f) v = beta * (*p) + v;
                *p = v;
            }
        }
    }
}

// =====================================================================
// helper kernels (fp32 VALU work)
// =====================================================================
__global__ void k_f32_to_f16_flat(const float* __restrict__ in,
                                  _Float16* __restrict__ out, long n)
{
    long i = (long)blockIdx.x * 256 + threadIdx.x;
    if (i < n) out[i] = (_Float16)in[i];
}

// strided-row fp32 -> contiguous f16  (row length fixed = 256)
__global__ void k_f32_to_f16_rows(const float* __restrict__ in,
                                  _Float16* __restrict__ out,
                                  long total, int rowsPerBatch, long batchStride)
{
    long i = (long)blockIdx.x * 256 + threadIdx.x;
    if (i >= total) return;
    long row = i >> 8;
    int  c   = (int)(i & 255);
    long b   = row / rowsPerBatch;
    long rr  = row % rowsPerBatch;
    out[i] = (_Float16)in[b * batchStride + rr * 256 + c];
}

__global__ void k_softmax_fg(const float* __restrict__ gw, float* __restrict__ out)
{
    int l = threadIdx.x;
    if (l >= EEL) return;
    const float* g = gw + l * FGN;
    float mx = g[0];
    for (int i = 1; i < FGN; ++i) mx = fmaxf(mx, g[i]);
    float e[FGN], s = 0.f;
    for (int i = 0; i < FGN; ++i) { e[i] = expf(g[i] - mx); s += e[i]; }
    for (int i = 0; i < FGN; ++i) out[l * FGN + i] = e[i] / s;
}

// forward DFT matrices (ortho): cosT/sinT are [Vp, L]; rows v>=V are zero
__global__ void k_build_dft_fwd(_Float16* __restrict__ cosT,
                                _Float16* __restrict__ sinT,
                                int L, int V, int Vp)
{
    long i = (long)blockIdx.x * 256 + threadIdx.x;
    if (i >= (long)Vp * L) return;
    int v = (int)(i / L), t = (int)(i % L);
    if (v >= V) { cosT[i] = (_Float16)0.0f; sinT[i] = (_Float16)0.0f; return; }
    long m = ((long)v * t) % L;
    float th = (float)(6.283185307179586 * (double)m / (double)L);
    float s, c; sincosf(th, &s, &c);
    float inv = rsqrtf((float)L);
    cosT[i] = (_Float16)(c * inv);
    sinT[i] = (_Float16)(-s * inv);
}

// inverse DFT matrices (ortho, hermitian doubling): ic/is are [L, Vp];
// columns v>=V are zero so K-sums over Vp are exact.
__global__ void k_build_dft_inv(_Float16* __restrict__ ic,
                                _Float16* __restrict__ isn,
                                int L, int V, int Vp)
{
    long i = (long)blockIdx.x * 256 + threadIdx.x;
    if (i >= (long)L * Vp) return;
    int t = (int)(i / Vp), v = (int)(i % Vp);
    if (v >= V) { ic[i] = (_Float16)0.0f; isn[i] = (_Float16)0.0f; return; }
    long m = ((long)v * t) % L;
    float th = (float)(6.283185307179586 * (double)m / (double)L);
    float s, c; sincosf(th, &s, &c);
    float a = (v == 0 || v == V - 1) ? 1.0f : 2.0f;   // V-1 == L/2 (Nyquist)
    float inv = rsqrtf((float)L);
    ic[i]  = (_Float16)(a * c * inv);
    isn[i] = (_Float16)(-a * s * inv);
}

__global__ void k_revin_stats(const float* __restrict__ x,
                              float* __restrict__ mean, float* __restrict__ stdv)
{
    int idx = blockIdx.x * 256 + threadIdx.x;       // B*C threads
    if (idx >= BB * CCH) return;
    int b = idx / CCH, c = idx % CCH;
    const float* p = x + (size_t)b * SSL * CCH + c;
    float s = 0.f;
    for (int t = 0; t < SSL; ++t) s += p[(size_t)t * CCH];
    float mu = s / SSL;
    float v = 0.f;
    for (int t = 0; t < SSL; ++t) { float d = p[(size_t)t * CCH] - mu; v += d * d; }
    v /= SSL;
    mean[idx] = mu;
    stdv[idx] = sqrtf(v) + 1e-5f;
}

__global__ void k_revin_apply(const float* __restrict__ x,
                              const float* __restrict__ mean,
                              const float* __restrict__ stdv,
                              const float* __restrict__ rw,
                              const float* __restrict__ rb,
                              float* __restrict__ z)
{
    long i = (long)blockIdx.x * 256 + threadIdx.x;  // B*S*C
    if (i >= (long)BB * SSL * CCH) return;
    long b = i / ((long)SSL * CCH);
    long rem = i - b * (long)SSL * CCH;
    long t = rem >> 8;
    int  c = (int)(rem & 255);
    float val = (x[i] - mean[b * CCH + c]) / stdv[b * CCH + c] * rw[c] + rb[c];
    z[(b * LL2 + t) * CCH + c] = val;               // z laid out [B, L2, C]
}

// layernorm over C=256; input rows can be strided, outputs contiguous f32+f16
__global__ __launch_bounds__(256)
void k_layernorm(const float* __restrict__ in, long batchStride,
                 int rowsPerBatch, long rowOffset,
                 const float* __restrict__ w, const float* __restrict__ bia,
                 float* __restrict__ out32, _Float16* __restrict__ out16)
{
    __shared__ float red[256];
    const int row = blockIdx.x;
    const int b   = row / rowsPerBatch;
    const int rr  = row % rowsPerBatch;
    const float* p = in + (size_t)b * batchStride + rowOffset + (size_t)rr * 256;
    const int t = threadIdx.x;
    float x = p[t];
    red[t] = x; __syncthreads();
    for (int s2 = 128; s2 > 0; s2 >>= 1) { if (t < s2) red[t] += red[t + s2]; __syncthreads(); }
    float mu = red[0] / 256.0f; __syncthreads();
    float d = x - mu;
    red[t] = d * d; __syncthreads();
    for (int s2 = 128; s2 > 0; s2 >>= 1) { if (t < s2) red[t] += red[t + s2]; __syncthreads(); }
    float var = red[0] / 256.0f;
    float y = d / sqrtf(var + 1e-5f) * w[t] + bia[t];
    out32[(size_t)row * 256 + t] = y;
    out16[(size_t)row * 256 + t] = (_Float16)y;
}

// Haar basis: out[m, i*GS+g] f16 from xg32[m,i], shift/scale_raw [C,GS]
__global__ void k_basis(const float* __restrict__ xg,
                        const float* __restrict__ shift,
                        const float* __restrict__ sraw,
                        _Float16* __restrict__ out, long total)
{
    long idx = (long)blockIdx.x * 256 + threadIdx.x;
    if (idx >= total) return;
    int  g = (int)(idx & 7);
    long t = idx >> 3;
    int  i = (int)(t & 255);
    long m = t >> 8;
    float x  = xg[(m << 8) + i];
    float sh = shift[(i << 3) + g];
    float sr = sraw[(i << 3) + g];
    float sp = (sr > 20.f) ? sr : log1pf(expf(sr));
    float sc = sp + 1e-8f;
    float gr = -1.0f + 0.25f * (float)g;
    float zz = (x - (sh + gr)) / sc;
    float wav = (zz >= 0.f && zz < 0.5f) ? 1.f : ((zz >= 0.5f && zz < 1.f) ? -1.f : 0.f);
    out[idx] = (_Float16)(wav * sqrtf(2.0f / sc));
}

__global__ void k_residual(float* __restrict__ z, const float* __restrict__ spec,
                           long total, int L)
{
    long i = (long)blockIdx.x * 256 + threadIdx.x;
    if (i >= total) return;
    long b = i / ((long)L * CCH);
    long rem = i - b * (long)L * CCH;
    z[b * (long)LL2 * CCH + rem] += spec[i];
}

__global__ void k_denorm(const float* __restrict__ dec,
                         const float* __restrict__ rw, const float* __restrict__ rb,
                         const float* __restrict__ mean, const float* __restrict__ stdv,
                         float* __restrict__ out)
{
    long i = (long)blockIdx.x * 256 + threadIdx.x;  // B*P*C
    if (i >= (long)BB * PPL * CCH) return;
    long b = i / ((long)PPL * CCH);
    long rem = i - b * (long)PPL * CCH;
    long p = rem >> 8;
    int  c = (int)(rem & 255);
    float d = dec[((size_t)b * LL2 + SSL + p) * CCH + c];
    out[i] = (d - rb[c]) / (rw[c] + 1e-5f) * stdv[b * CCH + c] + mean[b * CCH + c];
}

// =====================================================================
// host orchestration
// =====================================================================
static inline unsigned eb(long n) { return (unsigned)((n + 255) / 256); }

extern "C" void kernel_launch(void* const* d_in, const int* in_sizes, int n_in,
                              void* d_out, int out_size, void* d_ws, size_t ws_size,
                              hipStream_t stream)
{
    (void)in_sizes; (void)n_in; (void)out_size; (void)ws_size;
    const float* x_enc      = (const float*)d_in[0];
    const float* revin_w    = (const float*)d_in[1];
    const float* revin_b    = (const float*)d_in[2];
    const float* enc_norm_w = (const float*)d_in[3];
    const float* enc_norm_b = (const float*)d_in[4];
    const float* group_wts  = (const float*)d_in[5];
    const float* gn_w       = (const float*)d_in[6];
    const float* gn_b       = (const float*)d_in[7];
    const float* base_w     = (const float*)d_in[8];
    const float* haar_w     = (const float*)d_in[9];
    const float* haar_shift = (const float*)d_in[10];
    const float* haar_scale = (const float*)d_in[11];
    const float* pred_w     = (const float*)d_in[12];
    const float* pred_b     = (const float*)d_in[13];
    const float* proj_w     = (const float*)d_in[14];
    const float* proj_b     = (const float*)d_in[15];
    float* out = (float*)d_out;

    // ---- bump allocator over workspace ----
    char* wsb = (char*)d_ws;
    size_t off = 0;
    auto take = [&](size_t bytes) -> char* {
        char* p = wsb + off;
        off += (bytes + 255) & ~(size_t)255;
        return p;
    };
    const long nBase = (long)EEL * 2 * FGN * CCH * CCH;          // 1,048,576
    const long nHaar = nBase * GSN;                               // 8,388,608
    const long nPred = (long)LL2 * SSL;
    const long nProj = (long)CCH * CCH;
    const long nF1   = (long)VP1 * SSL;                           // padded
    const long nF2   = (long)VP2 * LL2;
    const long zEl   = (long)BB * LL2 * CCH;
    const long fqEl  = (long)BB * VP2 * CCH;
    const long xgEl  = (long)BB * VGMAX * CCH + 64 * CCH;         // +64 pad rows
    const long bsEl  = ((long)BB * VGMAX + 64) * CCH * GSN;       // +64 pad rows

    _Float16* w_base16 = (_Float16*)take(nBase * 2);
    _Float16* w_haar16 = (_Float16*)take(nHaar * 2);
    _Float16* w_pred16 = (_Float16*)take(nPred * 2);
    _Float16* w_proj16 = (_Float16*)take(nProj * 2);
    _Float16* cosT1 = (_Float16*)take(nF1 * 2);
    _Float16* sinT1 = (_Float16*)take(nF1 * 2);
    _Float16* ic1   = (_Float16*)take(nF1 * 2);
    _Float16* is1   = (_Float16*)take(nF1 * 2);
    _Float16* cosT2 = (_Float16*)take(nF2 * 2);
    _Float16* sinT2 = (_Float16*)take(nF2 * 2);
    _Float16* ic2   = (_Float16*)take(nF2 * 2);
    _Float16* is2   = (_Float16*)take(nF2 * 2);
    float* wsoft  = (float*)take(EEL * FGN * 4);
    float* msMean = (float*)take((long)BB * CCH * 4);
    float* msStd  = (float*)take((long)BB * CCH * 4);
    float* zA     = (float*)take(zEl * 4);
    float* zB     = (float*)take(zEl * 4);
    float* zn32   = (float*)take(zEl * 4);
    float* spec   = (float*)take(zEl * 4);     // also reused as `dec`
    float* freqR  = (float*)take(fqEl * 4);
    float* freqI  = (float*)take(fqEl * 4);
    float* xg32   = (float*)take(xgEl * 4);
    _Float16* zn16    = (_Float16*)take(zEl * 2);
    _Float16* z16     = (_Float16*)take(zEl * 2);
    _Float16* freqR16 = (_Float16*)take(fqEl * 2);
    _Float16* freqI16 = (_Float16*)take(fqEl * 2);
    _Float16* xg16    = (_Float16*)take(xgEl * 2);
    _Float16* basis16 = (_Float16*)take(bsEl * 2);

    auto gemm = [&](const _Float16* A, long sA, const _Float16* Bm, long sB,
                    float* Cm, long sC, int M, int N, int K, int b_kn,
                    const float* alpha, float beta,
                    const float* bias, int bias_mode, int batches) {
        dim3 grid((unsigned)((N + 63) / 64), (unsigned)((M + 63) / 64), (unsigned)batches);
        gemm_f16_wmma<<<grid, 256, 0, stream>>>(A, sA, Bm, sB, Cm, sC,
                                                M, N, K, b_kn, alpha, beta,
                                                bias, bias_mode);
    };
    auto cvt = [&](const float* in, _Float16* o, long n) {
        k_f32_to_f16_flat<<<eb(n), 256, 0, stream>>>(in, o, n);
    };

    // ---- one-time weight / matrix prep ----
    cvt(base_w, w_base16, nBase);
    cvt(haar_w, w_haar16, nHaar);
    cvt(pred_w, w_pred16, nPred);
    cvt(proj_w, w_proj16, nProj);
    k_softmax_fg<<<1, 32, 0, stream>>>(group_wts, wsoft);
    k_build_dft_fwd<<<eb(nF1), 256, 0, stream>>>(cosT1, sinT1, SSL, VV1, VP1);
    k_build_dft_inv<<<eb(nF1), 256, 0, stream>>>(ic1, is1, SSL, VV1, VP1);
    k_build_dft_fwd<<<eb(nF2), 256, 0, stream>>>(cosT2, sinT2, LL2, VV2, VP2);
    k_build_dft_inv<<<eb(nF2), 256, 0, stream>>>(ic2, is2, LL2, VV2, VP2);

    // ---- RevIN normalize ----
    k_revin_stats<<<eb((long)BB * CCH), 256, 0, stream>>>(x_enc, msMean, msStd);
    k_revin_apply<<<eb((long)BB * SSL * CCH), 256, 0, stream>>>(
        x_enc, msMean, msStd, revin_w, revin_b, zA);

    // ---- encoder (two passes, two layers each) ----
    //   V  = true rfft bin count (group split boundaries)
    //   Vp = padded bin count (layouts, GEMM M/K); pad rows are exact zeros
    auto encode = [&](float* z, int L, int V, int Vp,
                      const _Float16* cT, const _Float16* sT,
                      const _Float16* icM, const _Float16* isM) {
        for (int l = 0; l < EEL; ++l) {
            // layernorm over channels -> zn32/zn16 (contiguous [B*L,256])
            k_layernorm<<<BB * L, 256, 0, stream>>>(
                z, (long)LL2 * CCH, L, 0L,
                enc_norm_w + l * CCH, enc_norm_b + l * CCH, zn32, zn16);
            // rfft as GEMM: freq[Vp,C] = DFT[Vp,L] x zn[L,C]  (per batch)
            gemm(cT, 0, zn16, (long)L * CCH, freqR, (long)Vp * CCH,
                 Vp, CCH, L, 1, nullptr, 0.f, nullptr, 0, BB);
            gemm(sT, 0, zn16, (long)L * CCH, freqI, (long)Vp * CCH,
                 Vp, CCH, L, 1, nullptr, 0.f, nullptr, 0, BB);
            // per real/imag part, per frequency group
            for (int part = 0; part < 2; ++part) {
                float* fq = part ? freqI : freqR;
                for (int i = 0; i < FGN; ++i) {
                    int s = (i * V) / FGN;
                    int e = (i < FGN - 1) ? ((i + 1) * V) / FGN : V;
                    int Vg = e - s;
                    long pidx = ((long)(l * 2 + part) * FGN + i);
                    // group layernorm (reads fq slice) -> xg32/xg16
                    k_layernorm<<<BB * Vg, 256, 0, stream>>>(
                        fq, (long)Vp * CCH, Vg, (long)s * CCH,
                        gn_w + pidx * CCH, gn_b + pidx * CCH, xg32, xg16);
                    // Haar basis f16 [B*Vg, C*GS]
                    long nb = (long)BB * Vg * CCH * GSN;
                    k_basis<<<eb(nb), 256, 0, stream>>>(
                        xg32, haar_shift + pidx * CCH * GSN,
                        haar_scale + pidx * CCH * GSN, basis16, nb);
                    const float* alpha = wsoft + l * FGN + i;
                    // base: fq[s:e] = w * xg @ base_w^T
                    gemm(xg16, (long)Vg * CCH,
                         w_base16 + (size_t)pidx * CCH * CCH, 0,
                         fq + (long)s * CCH, (long)Vp * CCH,
                         Vg, CCH, CCH, 0, alpha, 0.f, nullptr, 0, BB);
                    // haar: fq[s:e] += w * basis @ haar_w^T  (K = 2048)
                    gemm(basis16, (long)Vg * CCH * GSN,
                         w_haar16 + (size_t)pidx * CCH * CCH * GSN, 0,
                         fq + (long)s * CCH, (long)Vp * CCH,
                         Vg, CCH, CCH * GSN, 0, alpha, 1.f, nullptr, 0, BB);
                }
            }
            // irfft as GEMM: spec[L,C] = IC[L,Vp] x R + IS[L,Vp] x I
            cvt(freqR, freqR16, (long)BB * Vp * CCH);
            cvt(freqI, freqI16, (long)BB * Vp * CCH);
            gemm(icM, 0, freqR16, (long)Vp * CCH, spec, (long)L * CCH,
                 L, CCH, Vp, 1, nullptr, 0.f, nullptr, 0, BB);
            gemm(isM, 0, freqI16, (long)Vp * CCH, spec, (long)L * CCH,
                 L, CCH, Vp, 1, nullptr, 1.f, nullptr, 0, BB);
            // residual
            k_residual<<<eb((long)BB * L * CCH), 256, 0, stream>>>(
                z, spec, (long)BB * L * CCH, L);
        }
    };

    encode(zA, SSL, VV1, VP1, cosT1, sinT1, ic1, is1);

    // ---- pred: zB[b, 0:768, c] = pred_w[768,512] x zA[b, 0:512, c] + pred_b ----
    k_f32_to_f16_rows<<<eb((long)BB * SSL * CCH), 256, 0, stream>>>(
        zA, z16, (long)BB * SSL * CCH, SSL, (long)LL2 * CCH);
    gemm(w_pred16, 0, z16, (long)SSL * CCH, zB, (long)LL2 * CCH,
         LL2, CCH, SSL, 1, nullptr, 0.f, pred_b, 2, BB);

    encode(zB, LL2, VV2, VP2, cosT2, sinT2, ic2, is2);

    // ---- proj: dec = zB @ proj_w^T + proj_b   (spec reused as dec) ----
    cvt(zB, z16, zEl);
    gemm(z16, 0, w_proj16, 0, spec, 0,
         BB * LL2, CCH, CCH, 0, nullptr, 0.f, proj_b, 1, 1);

    // ---- RevIN denorm of last P steps ----
    k_denorm<<<eb((long)BB * PPL * CCH), 256, 0, stream>>>(
        spec, revin_w, revin_b, msMean, msStd, out);
}
